// MoELayer_79353815761472
// MI455X (gfx1250) — compile-verified, matrix operand order checked by reading, not run on hardware
//
#include <hip/hip_runtime.h>

#define Bdim 4
#define Tdim 2048
#define Cdim 1024
#define Edim 8
#define Hdim 2048
#define Ntok (Bdim * Tdim)            // 8192
#define CAP  (2 * Ntok * 2 / Edim)    // 4096 tokens per expert
#define MT   32                       // M rows per block (two 16-row WMMA subtiles)

typedef __attribute__((ext_vector_type(16))) __bf16 v16bf;
typedef __attribute__((ext_vector_type(8)))  __bf16 v8bf;
typedef __attribute__((ext_vector_type(8)))  float  v8f;

static __device__ __forceinline__ unsigned short f2bf(float f) {
    unsigned int u = __builtin_bit_cast(unsigned int, f);
    unsigned int r = u + 0x7FFFu + ((u >> 16) & 1u);   // round-to-nearest-even
    return (unsigned short)(r >> 16);
}

static __device__ __forceinline__ v16bf combine(v8bf lo, v8bf hi) {
    v16bf r;
#pragma unroll
    for (int i = 0; i < 8; ++i) { r[i] = lo[i]; r[i + 8] = hi[i]; }
    return r;
}

static __device__ __forceinline__ float relu2(float v) {
    return v > 0.0f ? v * v : 0.0f;
}

// ---------------------------------------------------------------- init / cvt
__global__ void zero_f32_kernel(float* __restrict__ p, int n) {
    for (int i = blockIdx.x * blockDim.x + threadIdx.x; i < n; i += gridDim.x * blockDim.x)
        p[i] = 0.0f;
}

__global__ void zero_i32_kernel(int* __restrict__ p, int n) {
    for (int i = blockIdx.x * blockDim.x + threadIdx.x; i < n; i += gridDim.x * blockDim.x)
        p[i] = 0;
}

__global__ void cvt_bf16_kernel(const float* __restrict__ src, unsigned short* __restrict__ dst, int n) {
    for (int i = blockIdx.x * blockDim.x + threadIdx.x; i < n; i += gridDim.x * blockDim.x)
        dst[i] = f2bf(src[i]);
}

// ---------------------------------------------------------------- router
// one wave32 per token: scores -> softmax -> top-2 -> capacity slot assignment
__global__ __launch_bounds__(128)
void router_kernel(const float* __restrict__ x,          // [N, C]
                   const float* __restrict__ rw,         // [E, C]
                   float* __restrict__ route_out,        // [N, E]
                   int* __restrict__ counts,             // [E]
                   int* __restrict__ slot_token,         // [E*CAP]
                   float* __restrict__ slot_weight)      // [E*CAP]
{
    const int lane = threadIdx.x & 31;
    const int wid  = threadIdx.x >> 5;
    const int n    = blockIdx.x * 4 + wid;
    if (n >= Ntok) return;

    const float4* xrow = (const float4*)(x + (size_t)n * Cdim);
    float sc[Edim];
#pragma unroll
    for (int e = 0; e < Edim; ++e) {
        const float4* wr = (const float4*)(rw + (size_t)e * Cdim);
        float p = 0.0f;
        for (int i = lane; i < Cdim / 4; i += 32) {
            float4 a = xrow[i], b = wr[i];
            p += a.x * b.x + a.y * b.y + a.z * b.z + a.w * b.w;
        }
#pragma unroll
        for (int off = 16; off > 0; off >>= 1) p += __shfl_xor(p, off, 32);
        sc[e] = p;
    }
    // softmax over 8 (redundant on all lanes)
    float m = sc[0];
#pragma unroll
    for (int e = 1; e < Edim; ++e) m = fmaxf(m, sc[e]);
    float pr[Edim], s = 0.0f;
#pragma unroll
    for (int e = 0; e < Edim; ++e) { pr[e] = __expf(sc[e] - m); s += pr[e]; }
    const float inv = 1.0f / s;
#pragma unroll
    for (int e = 0; e < Edim; ++e) pr[e] *= inv;

    if (lane == 0) {
#pragma unroll
        for (int e = 0; e < Edim; ++e) route_out[(size_t)n * Edim + e] = pr[e];
        int e0 = 0;
#pragma unroll
        for (int e = 1; e < Edim; ++e) if (pr[e] > pr[e0]) e0 = e;
        int e1 = (e0 == 0) ? 1 : 0;
#pragma unroll
        for (int e = 0; e < Edim; ++e) if (e != e0 && pr[e] > pr[e1]) e1 = e;
        float w0 = pr[e0], w1 = pr[e1];
        const float rs = 1.0f / (w0 + w1 + 1e-10f);
        w0 *= rs; w1 *= rs;
        int p0 = atomicAdd(&counts[e0], 1);
        if (p0 < CAP) { slot_token[e0 * CAP + p0] = n; slot_weight[e0 * CAP + p0] = w0; }
        int p1 = atomicAdd(&counts[e1], 1);
        if (p1 < CAP) { slot_token[e1 * CAP + p1] = n; slot_weight[e1 * CAP + p1] = w1; }
    }
}

// ---------------------------------------------------------------- fused expert FFN
// block = (expert, 32-slot tile); 16 waves; one B tile feeds two M-subtile WMMAs.
// LDS: hidden 128KB + x 64KB (fits 320KB WGP budget)
__global__ __launch_bounds__(512)
void moe_expert_kernel(const unsigned short* __restrict__ xbf,   // [N, C] bf16
                       const unsigned short* __restrict__ fcwb,  // [E, H, C] bf16
                       const unsigned short* __restrict__ pwb,   // [E, C, H] bf16
                       const int* __restrict__ counts,
                       const int* __restrict__ slot_token,
                       const float* __restrict__ slot_weight,
                       float* __restrict__ out)                   // [N, C]
{
    extern __shared__ unsigned char smem[];
    __bf16* lds_h   = (__bf16*)smem;                  // [32][2048]  131072 B
    __bf16* lds_x   = (__bf16*)(smem + 131072);       // [32][1024]   65536 B
    int*    lds_tok = (int*)  (smem + 196608);        // [32]
    float*  lds_w   = (float*)(smem + 196736);        // [32]

    const int e    = blockIdx.x / (CAP / MT);
    const int tile = blockIdx.x % (CAP / MT);
    int cnt = counts[e]; if (cnt > CAP) cnt = CAP;
    if (tile * MT >= cnt) return;                     // prune empty capacity tiles
    const int slot0 = e * CAP + tile * MT;

    const int tid = threadIdx.x;
    if (tid < MT) {
        lds_tok[tid] = slot_token[slot0 + tid];       // unassigned -> token 0, weight 0
        lds_w[tid]   = slot_weight[slot0 + tid];
    }
    __syncthreads();

    // gather 32 token rows of x (bf16) into LDS, 16B chunks
    for (int ci = tid; ci < MT * 128; ci += 512) {
        const int r = ci >> 7, c8 = (ci & 127) << 3;
        *(v8bf*)(lds_x + r * Cdim + c8) =
            *(const v8bf*)(xbf + (size_t)lds_tok[r] * Cdim + c8);
    }
    __syncthreads();

    const int lane  = tid & 31;
    const int wave  = tid >> 5;                        // 0..15
    const int brow  = lane & 15;
    const int khalf = (lane >> 4) << 3;                // 0 | 8 (A/B lane K-chunk select)
    const int mhalf = (lane >> 4) << 3;                // C/D: lanes>=16 hold M+8

    // -------- Phase 1: hidden[32, H] = relu^2( X @ fc_w^T ); wave owns 128 H cols
    for (int nt = 0; nt < 8; ++nt) {
        const int nbase = wave * 128 + nt * 16;
        const unsigned short* bp = fcwb + ((size_t)e * Hdim + nbase + brow) * Cdim;
        __builtin_prefetch(bp, 0, 0);
        v8f acc0 = {};                                 // rows 0..15
        v8f acc1 = {};                                 // rows 16..31
#pragma unroll 2
        for (int k = 0; k < Cdim; k += 32) {
            v16bf b  = combine(*(const v8bf*)(bp + k + khalf),
                               *(const v8bf*)(bp + k + 16 + khalf));
            v16bf a0 = combine(*(const v8bf*)(lds_x + brow * Cdim + k + khalf),
                               *(const v8bf*)(lds_x + brow * Cdim + k + 16 + khalf));
            v16bf a1 = combine(*(const v8bf*)(lds_x + (16 + brow) * Cdim + k + khalf),
                               *(const v8bf*)(lds_x + (16 + brow) * Cdim + k + 16 + khalf));
            acc0 = __builtin_amdgcn_wmma_f32_16x16x32_bf16(false, a0, false, b,
                                                           (short)0, acc0, false, false);
            acc1 = __builtin_amdgcn_wmma_f32_16x16x32_bf16(false, a1, false, b,
                                                           (short)0, acc1, false, false);
        }
        const int ncol = nbase + brow;
        unsigned short* hs = (unsigned short*)lds_h;
#pragma unroll
        for (int i = 0; i < 8; ++i) {
            const int m = i + mhalf;
            hs[m * Hdim + ncol]        = f2bf(relu2(acc0[i]));
            hs[(16 + m) * Hdim + ncol] = f2bf(relu2(acc1[i]));
        }
    }
    __syncthreads();

    // -------- Phase 2: out[32, C] = hidden @ proj_w^T; wave owns 64 C cols
    for (int nt = 0; nt < 4; ++nt) {
        const int cbase = wave * 64 + nt * 16;
        const unsigned short* bp = pwb + ((size_t)e * Cdim + cbase + brow) * Hdim;
        __builtin_prefetch(bp, 0, 0);
        v8f acc0 = {};
        v8f acc1 = {};
#pragma unroll 2
        for (int k = 0; k < Hdim; k += 32) {
            v16bf b  = combine(*(const v8bf*)(bp + k + khalf),
                               *(const v8bf*)(bp + k + 16 + khalf));
            v16bf a0 = combine(*(const v8bf*)(lds_h + brow * Hdim + k + khalf),
                               *(const v8bf*)(lds_h + brow * Hdim + k + 16 + khalf));
            v16bf a1 = combine(*(const v8bf*)(lds_h + (16 + brow) * Hdim + k + khalf),
                               *(const v8bf*)(lds_h + (16 + brow) * Hdim + k + 16 + khalf));
            acc0 = __builtin_amdgcn_wmma_f32_16x16x32_bf16(false, a0, false, b,
                                                           (short)0, acc0, false, false);
            acc1 = __builtin_amdgcn_wmma_f32_16x16x32_bf16(false, a1, false, b,
                                                           (short)0, acc1, false, false);
        }
        const int c = cbase + brow;
#pragma unroll
        for (int i = 0; i < 8; ++i) {
            const int m = i + mhalf;
            atomicAdd(out + (size_t)lds_tok[m] * Cdim + c,      acc0[i] * lds_w[m]);
            atomicAdd(out + (size_t)lds_tok[16 + m] * Cdim + c, acc1[i] * lds_w[16 + m]);
        }
    }
}

// ---------------------------------------------------------------- launch
extern "C" void kernel_launch(void* const* d_in, const int* in_sizes, int n_in,
                              void* d_out, int out_size, void* d_ws, size_t ws_size,
                              hipStream_t stream) {
    const float* x   = (const float*)d_in[0];   // [B,T,C]
    const float* rw  = (const float*)d_in[1];   // [E,C]
    const float* fcw = (const float*)d_in[2];   // [E,H,C]
    const float* pw  = (const float*)d_in[3];   // [E,C,H]

    float* out       = (float*)d_out;                       // [N*C]
    float* route_out = out + (size_t)Ntok * Cdim;           // [N*E]

    unsigned char* ws = (unsigned char*)d_ws;
    unsigned short* xbf   = (unsigned short*)(ws);                // 16 MB
    unsigned short* fcwb  = (unsigned short*)(ws + 16777216);     // 32 MB
    unsigned short* pwb   = (unsigned short*)(ws + 50331648);     // 32 MB
    int*   counts      = (int*)(ws + 83886080);                   // [8]
    int*   slot_token  = counts + Edim;                           // [E*CAP]
    float* slot_weight = (float*)(slot_token + Edim * CAP);       // [E*CAP]

    // init (output must be zeroed every call; slot tables default token 0 / weight 0)
    zero_f32_kernel<<<2048, 256, 0, stream>>>(out, Ntok * Cdim);
    zero_i32_kernel<<<256, 256, 0, stream>>>(counts, Edim + 2 * Edim * CAP);

    // f32 -> bf16 staging
    cvt_bf16_kernel<<<2048, 256, 0, stream>>>(x,   xbf,  Ntok * Cdim);
    cvt_bf16_kernel<<<4096, 256, 0, stream>>>(fcw, fcwb, Edim * Hdim * Cdim);
    cvt_bf16_kernel<<<4096, 256, 0, stream>>>(pw,  pwb,  Edim * Cdim * Hdim);

    // router + slot assignment
    router_kernel<<<Ntok / 4, 128, 0, stream>>>(x, rw, route_out,
                                                counts, slot_token, slot_weight);

    // fused expert FFN (grid over expert x 32-row capacity tiles; empty tiles exit)
    moe_expert_kernel<<<Edim * (CAP / MT), 512, 196864, stream>>>(
        xbf, fcwb, pwb, counts, slot_token, slot_weight, out);
}
